// CategoricalGraphAtt_19456201851241
// MI455X (gfx1250) — matrix operand order repaired and checked by reading, not compiled, not run on hardware
//
#include <hip/hip_runtime.h>
#include <math.h>

// ---------------------------------------------------------------------------
// CategoricalGraphAtt for MI455X (gfx1250, wave32, WMMA)
// All dense GEMMs use v_wmma_f32_16x16x32_f16 (f16 in, f32 accum).
// ---------------------------------------------------------------------------

typedef __attribute__((ext_vector_type(16))) _Float16 v16h;
typedef __attribute__((ext_vector_type(8)))  _Float16 v8h;
typedef __attribute__((ext_vector_type(8)))  float    v8f;

constexpr int N_   = 20000;
constexpr int T_   = 10;
constexpr int D_   = 16;
constexpr int H_   = 128;
constexpr int WK_  = 3;
constexpr int S_   = 40;
constexpr int L_   = 500;
constexpr int EIN_ = 640000;
constexpr int EOUT_= 1600;
constexpr int G_   = 384;   // 3*H

#define DEVINL __device__ __forceinline__

DEVINL v8f bcast8(float v) {
  v8f r;
#pragma unroll
  for (int i = 0; i < 8; ++i) r[i] = v;
  return r;
}

DEVINL v8f wmma_f16(v16h a, v16h b, v8f c) {
  // D = A(16x32 f16) * B(32x16 f16) + C(16x16 f32)
  return __builtin_amdgcn_wmma_f32_16x16x32_f16(false, a, false, b, (short)0, c,
                                                false, false);
}

// A fragment (16x32) from f16 LDS tile, row-major with 'stride' halves per row.
// Layout: lanes 0-15 -> M=lane, halves0-7=K+0..7, halves8-15=K+16..23;
//         lanes 16-31 -> M=lane-16, halves0-7=K+8..15, halves8-15=K+24..31.
DEVINL v16h load_a_f16(const _Float16* base, int stride, int lane) {
  const int  ln = lane & 15;
  const bool hi = lane >= 16;
  const _Float16* p = base + (size_t)ln * stride + (hi ? 8 : 0);
  v8h a0 = *(const v8h*)(p);
  v8h a1 = *(const v8h*)(p + 16);
  v16h r;
#pragma unroll
  for (int i = 0; i < 8; ++i) { r[i] = a0[i]; r[8 + i] = a1[i]; }
  return r;
}

// Same A fragment but converting from an f32 LDS tile.
DEVINL v16h load_a_f32(const float* base, int stride, int lane) {
  const int  ln = lane & 15;
  const bool hi = lane >= 16;
  const float* p = base + (size_t)ln * stride + (hi ? 8 : 0);
  v16h r;
#pragma unroll
  for (int i = 0; i < 8; ++i) {
    r[i]     = (_Float16)p[i];
    r[8 + i] = (_Float16)p[16 + i];
  }
  return r;
}

// B fragment (32x16) from a weight stored row-major as W[outCol][K] (f16 LDS).
// wbase points at W[colBase + 0][kBase]. Layout: lane n=lane&15 -> col,
// lanes 0-15: halves0-7=K+0..7, halves8-15=K+16..23;
// lanes 16-31: halves0-7=K+8..15, halves8-15=K+24..31.
DEVINL v16h load_b_row(const _Float16* wbase, int stride, int lane) {
  const int  n  = lane & 15;
  const bool hi = lane >= 16;
  const _Float16* p = wbase + (size_t)n * stride + (hi ? 8 : 0);
  v8h b0 = *(const v8h*)(p);
  v8h b1 = *(const v8h*)(p + 16);
  v16h r;
#pragma unroll
  for (int i = 0; i < 8; ++i) { r[i] = b0[i]; r[8 + i] = b1[i]; }
  return r;
}

DEVINL float sigmoidf_(float x) { return 1.0f / (1.0f + __expf(-x)); }
DEVINL float lrelu02(float x)   { return x > 0.0f ? x : 0.2f * x; }
DEVINL unsigned fkey(float x) {
  unsigned u = __float_as_uint(x);
  return (u & 0x80000000u) ? ~u : (u | 0x80000000u);
}
DEVINL float funkey(unsigned u) {
  unsigned b = (u & 0x80000000u) ? (u & 0x7fffffffu) : ~u;
  return __uint_as_float(b);
}

// ---------------------------------------------------------------------------
// GRU over T=10 steps. Block owns 64 nodes; h lives in LDS across steps.
// Dynamic LDS layout (bytes):
//   [0)       Whh  f16 [384][128]   98304
//   [98304)   Wih  f16 [384][32]    24576  (K padded 16->32 with zeros)
//   [122880)  bih  f32 [384]         1536
//   [124416)  bhh  f32 [384]         1536
//   [125952)  h    f32 [64][128]    32768
//   [158720)  x_t  f16 [64][16]      2048   -> total 160768
// ---------------------------------------------------------------------------
__global__ __launch_bounds__(256) void gru_week_kernel(
    const float* __restrict__ x, const float* __restrict__ Wih,
    const float* __restrict__ Whh, const float* __restrict__ bih,
    const float* __restrict__ bhh, float* __restrict__ seq, int w) {
  extern __shared__ char smem[];
  _Float16* sWhh = (_Float16*)smem;
  _Float16* sWih = (_Float16*)(smem + 98304);
  float*    sbih = (float*)(smem + 122880);
  float*    sbhh = (float*)(smem + 124416);
  float*    sh   = (float*)(smem + 125952);
  _Float16* sx   = (_Float16*)(smem + 158720);

  const int  tid  = threadIdx.x;
  const int  lane = tid & 31;
  const int  wave = tid >> 5;
  const int  n0   = blockIdx.x * 64;
  const int  nact = min(64, N_ - n0);
  const int  mt   = wave & 3;          // M-tile (16 rows)
  const int  cg0  = (wave >> 2) * 4;   // first of 4 column-groups
  const bool act  = (mt * 16) < nact;
  const bool hi   = lane >= 16;
  const int  ln   = lane & 15;

  const float* Wihw = Wih + (size_t)w * G_ * D_;
  const float* Whhw = Whh + (size_t)w * G_ * H_;
  for (int i = tid; i < G_ * H_; i += 256) sWhh[i] = (_Float16)Whhw[i];
  // Wih zero-padded to K=32 so B fragments are plain vector LDS loads.
  for (int i = tid; i < G_ * 32; i += 256) {
    int g = i >> 5, k = i & 31;
    sWih[i] = (k < D_) ? (_Float16)Wihw[g * D_ + k] : (_Float16)0.0f;
  }
  for (int i = tid; i < G_; i += 256) { sbih[i] = bih[w * G_ + i]; sbhh[i] = bhh[w * G_ + i]; }
  for (int i = tid; i < 64 * H_; i += 256) sh[i] = 0.0f;
  __syncthreads();

  for (int t = 0; t < T_; ++t) {
    for (int i = tid; i < nact * D_; i += 256) {
      int nn = i >> 4, d = i & 15;
      sx[nn * D_ + d] = (_Float16)x[(((size_t)w * N_ + n0 + nn) * T_ + t) * D_ + d];
    }
    __syncthreads();

    float hnew[4][8];
    if (act) {
      // A fragment for x_t: K=16 zero-padded to 32
      v16h ax;
      {
        const _Float16* p = sx + (mt * 16 + ln) * D_ + (hi ? 8 : 0);
        v8h a0 = *(const v8h*)p;
#pragma unroll
        for (int i = 0; i < 8; ++i) { ax[i] = a0[i]; ax[8 + i] = (_Float16)0.0f; }
      }
      // A fragments for h: K=128 -> 4 chunks
      v16h ah[4];
#pragma unroll
      for (int kc = 0; kc < 4; ++kc)
        ah[kc] = load_a_f32(sh + (mt * 16) * H_ + kc * 32, H_, lane);

#pragma unroll
      for (int cg = 0; cg < 4; ++cg) {
        const int c = (cg0 + cg) * 16;
        // r and z gates only ever need xg+hg -> single accumulator chain
        v8f rz[2];
#pragma unroll
        for (int gi = 0; gi < 2; ++gi) {
          const int gb = gi * H_ + c;
          v8f acc = bcast8(sbih[gb + ln] + sbhh[gb + ln]);
          v16h bx = load_b_row(sWih + (size_t)gb * 32, 32, lane);
          acc = wmma_f16(ax, bx, acc);
#pragma unroll
          for (int kc = 0; kc < 4; ++kc) {
            v16h bh = load_b_row(sWhh + (size_t)gb * H_ + kc * 32, H_, lane);
            acc = wmma_f16(ah[kc], bh, acc);
          }
          rz[gi] = acc;
        }
        // n gate: xn and hn kept separate (r multiplies hn)
        const int gb = 2 * H_ + c;
        v8f xn = bcast8(sbih[gb + ln]);
        {
          v16h bx = load_b_row(sWih + (size_t)gb * 32, 32, lane);
          xn = wmma_f16(ax, bx, xn);
        }
        v8f hn = bcast8(sbhh[gb + ln]);
#pragma unroll
        for (int kc = 0; kc < 4; ++kc) {
          v16h bh = load_b_row(sWhh + (size_t)gb * H_ + kc * 32, H_, lane);
          hn = wmma_f16(ah[kc], bh, hn);
        }
#pragma unroll
        for (int r = 0; r < 8; ++r) {
          float rg = sigmoidf_(rz[0][r]);
          float zg = sigmoidf_(rz[1][r]);
          float ng = tanhf(xn[r] + rg * hn[r]);
          int row = mt * 16 + r + (hi ? 8 : 0);
          float hold = sh[row * H_ + c + ln];
          hnew[cg][r] = (1.0f - zg) * ng + zg * hold;
        }
      }
    }
    __syncthreads();
    if (act) {
#pragma unroll
      for (int cg = 0; cg < 4; ++cg) {
        const int c = (cg0 + cg) * 16;
#pragma unroll
        for (int r = 0; r < 8; ++r) {
          int row = mt * 16 + r + (hi ? 8 : 0);
          float v = hnew[cg][r];
          sh[row * H_ + c + ln] = v;
          seq[((size_t)(n0 + row) * T_ + t) * H_ + c + ln] = v;
        }
      }
    }
    __syncthreads();
  }
}

// ---------------------------------------------------------------------------
// Generic C[M,128] = A[M,K] @ W[128,K]^T (+bias)(ReLU), K in {128,384}.
// Dynamic LDS: W f16 [128][K] then A f16 [64][K] -> 384*K bytes.
// ---------------------------------------------------------------------------
__global__ __launch_bounds__(256) void gemm128_kernel(
    const float* __restrict__ A, const float* __restrict__ Wt,
    const float* __restrict__ bias, float* __restrict__ Cout,
    int M, int K, int relu) {
  extern __shared__ char smem[];
  _Float16* sW = (_Float16*)smem;
  _Float16* sA = (_Float16*)(smem + (size_t)128 * K * 2);
  const int  tid  = threadIdx.x;
  const int  lane = tid & 31;
  const int  wave = tid >> 5;
  const bool hi   = lane >= 16;
  const int  ln   = lane & 15;
  const int  n0   = blockIdx.x * 64;

  for (int i = tid; i < 128 * K; i += 256) sW[i] = (_Float16)Wt[i];
  for (int i = tid; i < 64 * K; i += 256) {
    int r = i / K, k = i - r * K;
    int row = n0 + r;
    sA[i] = (row < M) ? (_Float16)A[(size_t)row * K + k] : (_Float16)0.0f;
  }
  __syncthreads();

  const int mt  = wave & 3;
  const int cg0 = (wave >> 2) * 4;
#pragma unroll
  for (int cg = 0; cg < 4; ++cg) {
    const int c = (cg0 + cg) * 16;
    v8f acc = bias ? bcast8(bias[c + ln]) : bcast8(0.0f);
    for (int kc = 0; kc < K / 32; ++kc) {
      v16h a = load_a_f16(sA + (size_t)(mt * 16) * K + kc * 32, K, lane);
      v16h b = load_b_row(sW + (size_t)c * K + kc * 32, K, lane);
      acc = wmma_f16(a, b, acc);
    }
#pragma unroll
    for (int r = 0; r < 8; ++r) {
      int row = n0 + mt * 16 + r + (hi ? 8 : 0);
      if (row < M) {
        float v = acc[r];
        if (relu) v = fmaxf(v, 0.0f);
        Cout[(size_t)row * 128 + c + ln] = v;
      }
    }
  }
}

// ---------------------------------------------------------------------------
// Temporal attention pooling (T=10) per (node, channel).
// ---------------------------------------------------------------------------
__global__ void att_time_kernel(const float* __restrict__ seq,
                                const float* __restrict__ aW,
                                const float* __restrict__ ab,
                                float* __restrict__ wkemb, int w) {
  int idx = blockIdx.x * blockDim.x + threadIdx.x;
  if (idx >= N_ * H_) return;
  int n = idx >> 7, h = idx & 127;
  float sv[T_];
#pragma unroll
  for (int t = 0; t < T_; ++t) sv[t] = seq[((size_t)n * T_ + t) * H_ + h];
  float lg[T_];
  float mx = -1e30f;
#pragma unroll
  for (int tp = 0; tp < T_; ++tp) {
    float acc = ab[w * T_ + tp];
#pragma unroll
    for (int t = 0; t < T_; ++t) acc += aW[((size_t)w * T_ + tp) * T_ + t] * sv[t];
    lg[tp] = acc;
    mx = fmaxf(mx, acc);
  }
  float den = 0.0f, o = 0.0f;
#pragma unroll
  for (int tp = 0; tp < T_; ++tp) {
    float e = __expf(lg[tp] - mx);
    den += e;
    o += e * sv[tp];
  }
  wkemb[((size_t)n * WK_ + w) * H_ + h] = o / den;
}

__global__ void att_week_kernel(const float* __restrict__ wkemb,
                                const float* __restrict__ kW,
                                const float* __restrict__ kb,
                                float* __restrict__ wav) {
  int idx = blockIdx.x * blockDim.x + threadIdx.x;
  if (idx >= N_ * H_) return;
  int n = idx >> 7, h = idx & 127;
  float sv[WK_];
#pragma unroll
  for (int w = 0; w < WK_; ++w) sv[w] = wkemb[((size_t)n * WK_ + w) * H_ + h];
  float lg[WK_];
  float mx = -1e30f;
#pragma unroll
  for (int wp = 0; wp < WK_; ++wp) {
    float acc = kb[wp];
#pragma unroll
    for (int w = 0; w < WK_; ++w) acc += kW[wp * WK_ + w] * sv[w];
    lg[wp] = acc;
    mx = fmaxf(mx, acc);
  }
  float den = 0.0f, o = 0.0f;
#pragma unroll
  for (int wp = 0; wp < WK_; ++wp) {
    float e = __expf(lg[wp] - mx);
    den += e;
    o += e * sv[wp];
  }
  wav[(size_t)n * H_ + h] = o / den;
}

// ---------------------------------------------------------------------------
// GAT helpers (used for both graphs).
// ---------------------------------------------------------------------------
__global__ void gat_dots_kernel(const float* __restrict__ xp,
                                const float* __restrict__ asrc,
                                const float* __restrict__ adst,
                                float* __restrict__ es, float* __restrict__ ed, int n) {
  int i = blockIdx.x * blockDim.x + threadIdx.x;
  if (i >= n) return;
  const float* r = xp + (size_t)i * H_;
  float a = 0.0f, b = 0.0f;
  for (int k = 0; k < H_; ++k) { float v = r[k]; a += v * asrc[k]; b += v * adst[k]; }
  es[i] = a; ed[i] = b;
}

__global__ void gat_init_kernel(const float* es, const float* ed, unsigned* mkey, int n) {
  int i = blockIdx.x * blockDim.x + threadIdx.x;
  if (i >= n) return;
  mkey[i] = fkey(lrelu02(es[i] + ed[i]));   // self-loop seeds the max
}

__global__ void gat_edge_max_kernel(const int* __restrict__ eidx, const float* es,
                                    const float* ed, float* abuf, unsigned* mkey, int E) {
  int e = blockIdx.x * blockDim.x + threadIdx.x;
  if (e >= E) return;
  int s = eidx[e], d = eidx[E + e];
  float a = lrelu02(es[s] + ed[d]);
  abuf[e] = a;
  atomicMax(&mkey[d], fkey(a));
}

__global__ void gat_node_den_kernel(const float* es, const float* ed, const unsigned* mkey,
                                    float* m, float* den, float* exs, int n) {
  int i = blockIdx.x * blockDim.x + threadIdx.x;
  if (i >= n) return;
  float mm = funkey(mkey[i]);
  m[i] = mm;
  float e0 = __expf(lrelu02(es[i] + ed[i]) - mm);   // self-loop term
  den[i] = e0;
  exs[i] = e0;
}

__global__ void gat_edge_den_kernel(const int* __restrict__ eidx, const float* m,
                                    float* abuf, float* den, int E) {
  int e = blockIdx.x * blockDim.x + threadIdx.x;
  if (e >= E) return;
  int d = eidx[E + e];
  float ex = __expf(abuf[e] - m[d]);
  abuf[e] = ex;
  atomicAdd(&den[d], ex);
}

__global__ void gat_node_out_kernel(const float* __restrict__ xp, const float* exs,
                                    const float* den, const float* __restrict__ bias,
                                    float* __restrict__ out, int n) {
  int idx = blockIdx.x * blockDim.x + threadIdx.x;
  if (idx >= n * H_) return;
  int i = idx >> 7, h = idx & 127;
  out[idx] = (exs[i] / den[i]) * xp[idx] + bias[h];
}

__global__ void gat_edge_scatter_kernel(const int* __restrict__ eidx,
                                        const float* __restrict__ xp,
                                        const float* __restrict__ abuf,
                                        const float* __restrict__ den,
                                        float* __restrict__ out, int E) {
  int wave = threadIdx.x >> 5, lane = threadIdx.x & 31;
  int e = blockIdx.x * 8 + wave;
  if (e >= E) return;
  int s = eidx[e], d = eidx[E + e];
  float alpha = abuf[e] / den[d];
  const float* xr = xp + (size_t)s * H_ + lane * 4;
  float* orow = out + (size_t)d * H_ + lane * 4;
#pragma unroll
  for (int i = 0; i < 4; ++i) atomicAdd(&orow[i], alpha * xr[i]);
}

// ---------------------------------------------------------------------------
// Per-sector attention pooling: logits[s] = st_s[128,500] @ pool_W[s]^T,
// online softmax over t with butterfly reduction. One block per sector.
// B tiles staged in LDS per column-tile (block-synchronous ct loop).
// Dynamic LDS: st f16 [128][512] (131072) + bias f32 [512] (2048)
//              + Btile f16 [16][512] (16384) -> 149504 bytes.
// ---------------------------------------------------------------------------
__global__ __launch_bounds__(256) void sector_pool_kernel(
    const float* __restrict__ inner, const float* __restrict__ pW,
    const float* __restrict__ pb, float* __restrict__ cat) {
  extern __shared__ char smem[];
  _Float16* st = (_Float16*)smem;
  float*    bl = (float*)(smem + 131072);
  _Float16* sB = (_Float16*)(smem + 133120);   // [16 cols][512 k]
  const int  s    = blockIdx.x;
  const int  tid  = threadIdx.x;
  const int  lane = tid & 31;
  const int  wave = tid >> 5;     // 8 waves = 8 M-tiles of h
  const bool hi   = lane >= 16;
  const int  ln   = lane & 15;

  for (int i = tid; i < 128 * 512; i += 256) {
    int h = i >> 9, l = i & 511;
    st[i] = (l < L_) ? (_Float16)inner[((size_t)s * L_ + l) * H_ + h] : (_Float16)0.0f;
  }
  for (int i = tid; i < 512; i += 256) bl[i] = (i < L_) ? pb[s * L_ + i] : 0.0f;

  float m[8], z[8], a[8];
#pragma unroll
  for (int r = 0; r < 8; ++r) { m[r] = -1e30f; z[r] = 0.0f; a[r] = 0.0f; }

  for (int ct = 0; ct < 32; ++ct) {
    __syncthreads();
    // Stage B tile for columns [ct*16, ct*16+16): sB[tt][k] = pool_W[s][t][k]
    for (int i = tid; i < 16 * 512; i += 256) {
      int tt = i >> 9, k = i & 511;
      int tc = ct * 16 + tt;
      sB[i] = (tc < L_ && k < L_) ? (_Float16)pW[((size_t)s * L_ + tc) * L_ + k]
                                  : (_Float16)0.0f;
    }
    __syncthreads();

    const int  tcol = ct * 16 + ln;
    const bool tin  = tcol < L_;
    v8f acc = bcast8(0.0f);
    for (int kc = 0; kc < 16; ++kc) {
      v16h af = load_a_f16(st + (size_t)(wave * 16) * 512 + kc * 32, 512, lane);
      v16h bf = load_b_row(sB + kc * 32, 512, lane);
      acc = wmma_f16(af, bf, acc);
    }
    if (tin) {
      float bv = bl[tcol];
#pragma unroll
      for (int r = 0; r < 8; ++r) {
        int   hrow = wave * 16 + r + (hi ? 8 : 0);
        float lg = acc[r] + bv;
        float sv = (float)st[(size_t)hrow * 512 + tcol];
        float mn = fmaxf(m[r], lg);
        float sc = __expf(m[r] - mn);
        float ev = __expf(lg - mn);
        z[r] = z[r] * sc + ev;
        a[r] = a[r] * sc + ev * sv;
        m[r] = mn;
      }
    }
  }
  // reduce across the 16 columns each half-wave holds
#pragma unroll
  for (int off = 1; off < 16; off <<= 1) {
#pragma unroll
    for (int r = 0; r < 8; ++r) {
      float m2 = __shfl_xor(m[r], off, 32);
      float z2 = __shfl_xor(z[r], off, 32);
      float a2 = __shfl_xor(a[r], off, 32);
      float mn = fmaxf(m[r], m2);
      float s1 = __expf(m[r] - mn), s2 = __expf(m2 - mn);
      z[r] = z[r] * s1 + z2 * s2;
      a[r] = a[r] * s1 + a2 * s2;
      m[r] = mn;
    }
  }
  if (ln == 0) {
#pragma unroll
    for (int r = 0; r < 8; ++r) {
      int hrow = wave * 16 + r + (hi ? 8 : 0);
      cat[s * H_ + hrow] = a[r] / z[r];
    }
  }
}

// ---------------------------------------------------------------------------
// Fusion concat + output heads.
// ---------------------------------------------------------------------------
__global__ void concat_kernel(const float* __restrict__ wav,
                              const float* __restrict__ inner,
                              const float* __restrict__ cat2,
                              float* __restrict__ F) {
  int idx = blockIdx.x * blockDim.x + threadIdx.x;
  if (idx >= N_ * G_) return;
  int n = idx / G_, c = idx - n * G_;
  float v;
  if (c < 128)      v = wav[(size_t)n * H_ + c];
  else if (c < 256) v = inner[(size_t)n * H_ + c - 128];
  else              v = cat2[(size_t)(n / L_) * H_ + c - 256];
  F[idx] = v;
}

__global__ void heads_kernel(const float* __restrict__ fus,
                             const float* __restrict__ rW, const float* __restrict__ rb,
                             const float* __restrict__ cW, const float* __restrict__ cb,
                             float* __restrict__ out) {
  int i = blockIdx.x * blockDim.x + threadIdx.x;
  if (i >= N_) return;
  const float* f = fus + (size_t)i * H_;
  float r = rb[0], c = cb[0];
  for (int k = 0; k < H_; ++k) { r += f[k] * rW[k]; c += f[k] * cW[k]; }
  out[i]      = r;
  out[N_ + i] = sigmoidf_(c);
}

// ---------------------------------------------------------------------------
extern "C" void kernel_launch(void* const* d_in, const int* in_sizes, int n_in,
                              void* d_out, int out_size, void* d_ws, size_t ws_size,
                              hipStream_t stream) {
  (void)in_sizes; (void)n_in; (void)out_size; (void)ws_size;
  const float* x        = (const float*)d_in[0];
  const int*   in_edge  = (const int*)d_in[1];
  const int*   out_edge = (const int*)d_in[2];
  const float* gWih = (const float*)d_in[3];
  const float* gWhh = (const float*)d_in[4];
  const float* gbih = (const float*)d_in[5];
  const float* gbhh = (const float*)d_in[6];
  const float* attW = (const float*)d_in[7];
  const float* attb = (const float*)d_in[8];
  const float* wkW  = (const float*)d_in[9];
  const float* wkb  = (const float*)d_in[10];
  const float* g1W  = (const float*)d_in[11];
  const float* g1as = (const float*)d_in[12];
  const float* g1ad = (const float*)d_in[13];
  const float* g1b  = (const float*)d_in[14];
  const float* g2W  = (const float*)d_in[15];
  const float* g2as = (const float*)d_in[16];
  const float* g2ad = (const float*)d_in[17];
  const float* g2b  = (const float*)d_in[18];
  const float* pW   = (const float*)d_in[19];
  const float* pb   = (const float*)d_in[20];
  const float* fW   = (const float*)d_in[21];
  const float* fb   = (const float*)d_in[22];
  const float* rW   = (const float*)d_in[23];
  const float* rb   = (const float*)d_in[24];
  const float* cW   = (const float*)d_in[25];
  const float* cb   = (const float*)d_in[26];
  float* out = (float*)d_out;
  float* ws  = (float*)d_ws;

  // Workspace layout (floats). ~41.2M floats (~165 MB) total.
  size_t o = 0;
  float* seq   = ws;            o += (size_t)N_ * T_ * H_;   // 25.6M
  float* wkemb = ws + o;        o += (size_t)N_ * WK_ * H_;  // 7.68M
  float* wav   = ws + o;        o += (size_t)N_ * H_;
  float* xp1   = ws + o;        o += (size_t)N_ * H_;
  float* g1out = ws + o;        o += (size_t)N_ * H_;
  float* es1   = ws + o;        o += N_;
  float* ed1   = ws + o;        o += N_;
  unsigned* mk1 = (unsigned*)(ws + o); o += N_;
  float* mm1   = ws + o;        o += N_;
  float* den1  = ws + o;        o += N_;
  float* exs1  = ws + o;        o += N_;
  float* sec   = ws + o;        o += S_ * H_;
  float* xp2   = ws + o;        o += S_ * H_;
  float* g2out = ws + o;        o += S_ * H_;
  float* es2   = ws + o;        o += S_;
  float* ed2   = ws + o;        o += S_;
  unsigned* mk2 = (unsigned*)(ws + o); o += S_;
  float* mm2   = ws + o;        o += S_;
  float* den2  = ws + o;        o += S_;
  float* exs2  = ws + o;        o += S_;
  float* ae2   = ws + o;        o += EOUT_;
  // Aliased into the seq region (seq is dead after att_week; regions disjoint):
  float* F    = seq;                                      // N*384
  float* fus  = seq + (size_t)N_ * G_;                    // N*128
  float* ae1  = seq + (size_t)N_ * G_ + (size_t)N_ * H_;  // E_IN

  const int TPB = 256;
  const int tileBlocks = (N_ + 63) / 64;               // 313
  const int nhBlocks   = (N_ * H_ + TPB - 1) / TPB;
  const int nBlocks    = (N_ + TPB - 1) / TPB;

  // 1) per-week GRU + temporal attention pooling
  for (int w = 0; w < WK_; ++w) {
    gru_week_kernel<<<tileBlocks, TPB, 160768, stream>>>(x, gWih, gWhh, gbih, gbhh, seq, w);
    att_time_kernel<<<nhBlocks, TPB, 0, stream>>>(seq, attW, attb, wkemb, w);
  }
  // 2) weekly attention pooling
  att_week_kernel<<<nhBlocks, TPB, 0, stream>>>(wkemb, wkW, wkb, wav);

  // 3) GAT on inner graph
  gemm128_kernel<<<tileBlocks, TPB, (size_t)384 * H_, stream>>>(wav, g1W, nullptr, xp1, N_, H_, 0);
  gat_dots_kernel<<<nBlocks, TPB, 0, stream>>>(xp1, g1as, g1ad, es1, ed1, N_);
  gat_init_kernel<<<nBlocks, TPB, 0, stream>>>(es1, ed1, mk1, N_);
  const int eB = (EIN_ + TPB - 1) / TPB;
  gat_edge_max_kernel<<<eB, TPB, 0, stream>>>(in_edge, es1, ed1, ae1, mk1, EIN_);
  gat_node_den_kernel<<<nBlocks, TPB, 0, stream>>>(es1, ed1, mk1, mm1, den1, exs1, N_);
  gat_edge_den_kernel<<<eB, TPB, 0, stream>>>(in_edge, mm1, ae1, den1, EIN_);
  gat_node_out_kernel<<<nhBlocks, TPB, 0, stream>>>(xp1, exs1, den1, g1b, g1out, N_);
  gat_edge_scatter_kernel<<<(EIN_ + 7) / 8, TPB, 0, stream>>>(in_edge, xp1, ae1, den1, g1out, EIN_);

  // 4) per-sector attention pooling (WMMA batch GEMM + online softmax)
  sector_pool_kernel<<<S_, TPB, 149504, stream>>>(g1out, pW, pb, sec);

  // 5) GAT on outer graph
  gemm128_kernel<<<1, TPB, (size_t)384 * H_, stream>>>(sec, g2W, nullptr, xp2, S_, H_, 0);
  gat_dots_kernel<<<1, TPB, 0, stream>>>(xp2, g2as, g2ad, es2, ed2, S_);
  gat_init_kernel<<<1, TPB, 0, stream>>>(es2, ed2, mk2, S_);
  const int e2B = (EOUT_ + TPB - 1) / TPB;
  gat_edge_max_kernel<<<e2B, TPB, 0, stream>>>(out_edge, es2, ed2, ae2, mk2, EOUT_);
  gat_node_den_kernel<<<1, TPB, 0, stream>>>(es2, ed2, mk2, mm2, den2, exs2, S_);
  gat_edge_den_kernel<<<e2B, TPB, 0, stream>>>(out_edge, mm2, ae2, den2, EOUT_);
  gat_node_out_kernel<<<(S_ * H_ + TPB - 1) / TPB, TPB, 0, stream>>>(xp2, exs2, den2, g2b, g2out, S_);
  gat_edge_scatter_kernel<<<(EOUT_ + 7) / 8, TPB, 0, stream>>>(out_edge, xp2, ae2, den2, g2out, EOUT_);

  // 6) fusion (concat -> GEMM K=384 with bias+ReLU) and heads
  concat_kernel<<<(N_ * G_ + TPB - 1) / TPB, TPB, 0, stream>>>(wav, g1out, g2out, F);
  gemm128_kernel<<<tileBlocks, TPB, (size_t)384 * G_, stream>>>(F, fW, fb, fus, N_, G_, 1);
  heads_kernel<<<nBlocks, TPB, 0, stream>>>(fus, rW, rb, cW, cb, out);
}
